// UNet6_5085241279034
// MI455X (gfx1250) — compile-verified, matrix operand order checked by reading, not run on hardware
//
#include <hip/hip_runtime.h>
#include <cstdint>

// ---------------------------------------------------------------------------
// Types for CDNA5 WMMA (wave32): 16x16x32 bf16 -> f32
// ---------------------------------------------------------------------------
typedef __attribute__((ext_vector_type(16))) __bf16 v16bf;
typedef __attribute__((ext_vector_type(8)))  float  v8f;

union FragB { uint4 q[2]; v16bf v; };

__device__ __forceinline__ unsigned short f2bf(float f) {
  unsigned int u = __float_as_uint(f);
  u += 0x7FFFu + ((u >> 16) & 1u);          // round-to-nearest-even
  return (unsigned short)(u >> 16);
}

// ---------------------------------------------------------------------------
// BatchNorm statistics: two-pass (sum, sumsq) -> per-channel scale/shift
// ---------------------------------------------------------------------------
__global__ void zero_stats(float* __restrict__ s) { s[threadIdx.x] = 0.f; }

__global__ __launch_bounds__(256) void bn_sums(const float* __restrict__ x, int N,
                                               float* __restrict__ st) {
  __shared__ float ssum[256], ssq[256];
  const int c  = threadIdx.x & 63;
  const int sl = threadIdx.x >> 6;                 // 4 row-slices per block
  float s = 0.f, q = 0.f;
  for (int r = blockIdx.x * 4 + sl; r < N; r += gridDim.x * 4) {
    float v = x[(size_t)r * 64 + c];
    s += v; q += v * v;
  }
  ssum[threadIdx.x] = s; ssq[threadIdx.x] = q;
  __syncthreads();
  if (sl == 0) {
    for (int i = 1; i < 4; ++i) { s += ssum[i * 64 + c]; q += ssq[i * 64 + c]; }
    atomicAdd(&st[c], s);
    atomicAdd(&st[64 + c], q);
  }
}

__global__ void bn_fin(const float* __restrict__ st, const float* __restrict__ g,
                       const float* __restrict__ b, int N, float* __restrict__ ss) {
  int c = threadIdx.x;                             // 64 threads
  float inv = 1.0f / (float)N;
  float m   = st[c] * inv;
  float var = st[64 + c] * inv - m * m;
  float sc  = g[c] * rsqrtf(var + 1e-4f);
  ss[c]      = sc;
  ss[64 + c] = b[c] - m * sc;
}

// ---------------------------------------------------------------------------
// Fused BNReLU + fp32->bf16 pre-pass (once per conv, not per tap!).
// Writes y[N+1][64] bf16 with row N = zeros (the rulebook pad row).
// ---------------------------------------------------------------------------
__global__ __launch_bounds__(256) void bnrelu_bf16(const float* __restrict__ x, int N,
                                                   const float* __restrict__ ss,
                                                   unsigned short* __restrict__ y) {
  int i = blockIdx.x * 256 + threadIdx.x;
  if (i >= (N + 1) * 64) return;
  int r = i >> 6, c = i & 63;
  float v = 0.f;
  if (r < N) v = fmaxf(x[(size_t)r * 64 + c] * ss[c] + ss[64 + c], 0.f);
  y[i] = f2bf(v);
}

// ---------------------------------------------------------------------------
// Weight conversion: fp32 [taps][cin][cout] -> bf16 [taps][cout][cin]
// (transposed so B fragments are contiguous 16B global loads per lane)
// ---------------------------------------------------------------------------
__global__ __launch_bounds__(256) void cvt_w(const float* __restrict__ w, int taps,
                                             unsigned short* __restrict__ o) {
  int i = blockIdx.x * 256 + threadIdx.x;
  if (i >= taps * 4096) return;
  int k = i >> 12, rc = i & 4095, cin = rc >> 6, cout = rc & 63;
  o[(k << 12) + (cout << 6) + cin] = f2bf(w[i]);
}

// ---------------------------------------------------------------------------
// Deconv rulebook: tap k index = (offset[i]==k) ? parent[i] : PAD
// ---------------------------------------------------------------------------
__global__ __launch_bounds__(256) void build_up_idx(const int* __restrict__ parent,
                                                    const int* __restrict__ offset,
                                                    int M, int Ncoarse,
                                                    int* __restrict__ outIdx) {
  int i = blockIdx.x * 256 + threadIdx.x;
  if (i >= M) return;
  int p = parent[i], o = offset[i];
#pragma unroll
  for (int k = 0; k < 8; ++k) outIdx[(size_t)i * 8 + k] = (k == o) ? p : Ncoarse;
}

// ---------------------------------------------------------------------------
// Core: rulebook gather (async DMA to LDS, double-buffered over ASYNCcnt) +
// multi-tap GEMM via WMMA bf16.
// Block = 256 threads (8 waves). Tile = 32 sites x 64 out-channels.
// Wave w -> (mtile = w>>2, ntile = w&3) 16x16 WMMA tile, K = TAPS*64.
// Pipeline: at tap k, issue tap k+1's global_load_async_to_lds_b128 into the
// other LDS buffer, then s_wait_asynccnt 1 (in-order completion => tap k's
// tile landed) so the k+1 DMA overlaps tap k's WMMAs.
// B fragments straight from global (pre-transposed bf16 weights, L2-resident).
// ---------------------------------------------------------------------------
template <int TAPS>
__global__ __launch_bounds__(256) void gather_gemm_wmma(
    const unsigned short* __restrict__ y,        // [padrow+1][64] bf16, row padrow = 0
    int padrow,
    const int* __restrict__ idx, int M,
    const unsigned short* __restrict__ Wt,       // [TAPS][64(cout)][64(cin)] bf16
    const float* __restrict__ base, float* __restrict__ out) {
  __shared__ __align__(16) unsigned short As[2][32 * 72];  // 2 x 4608B tiles

  const int t    = threadIdx.x;
  const int lane = t & 31;
  const int wave = t >> 5;
  const int mt   = wave >> 2;     // 0..1
  const int nt   = wave & 3;      // 0..3
  const int row0 = blockIdx.x * 32;

  // gather assignment: 32 rows x 8 chunks of 16B (one async b128 per lane/tap)
  const int gr  = t >> 3;
  const int gch = t & 7;
  const int site = row0 + gr;
  // low 32 bits of the generic address of LDS == LDS byte offset
  const unsigned laddr0 =
      (unsigned)(uintptr_t)(&As[0][0]) + (unsigned)(gr * 144 + gch * 16);

  const int m  = lane & 15;
  const int hf = lane >> 4;
  const unsigned short* apb = &As[0][(mt * 16 + m) * 72 + hf * 8];

  v8f acc = {};

  auto issue = [&](int k, int buf) {
    int j = (site < M) ? idx[(size_t)site * TAPS + k] : padrow;
    const unsigned short* gp = y + (size_t)j * 64 + gch * 8;
    unsigned la = laddr0 + (unsigned)(buf * 4608);
    asm volatile("global_load_async_to_lds_b128 %0, %1, off"
                 :: "v"(la), "v"(gp) : "memory");
  };
  auto compute = [&](int k, int buf) {
#pragma unroll
    for (int ks = 0; ks < 2; ++ks) {             // 2 x K=32 per tap
      FragB a, b;
      // A 16x32 bf16 (ISA 7.12.2): lane half selects K-octet interleave
      const unsigned short* ap = apb + buf * 2304 + ks * 32;
      a.q[0] = *(const uint4*)ap;                // K = 8h .. 8h+7
      a.q[1] = *(const uint4*)(ap + 16);         // K = 16+8h .. 16+8h+7
      // B 32x16 bf16: lane -> col (cout); weights pre-transposed [cout][cin]
      const unsigned short* wp =
          Wt + ((size_t)k * 64 + nt * 16 + m) * 64 + ks * 32 + hf * 8;
      b.q[0] = *(const uint4*)wp;
      b.q[1] = *(const uint4*)(wp + 16);
      acc = __builtin_amdgcn_wmma_f32_16x16x32_bf16(false, a.v, false, b.v,
                                                    (short)0, acc, false, false);
    }
  };

  issue(0, 0);                                   // prologue: tap 0 DMA in flight
#pragma unroll 1
  for (int k = 0; k < TAPS - 1; ++k) {
    issue(k + 1, (k + 1) & 1);                   // overlap next DMA with compute
    __builtin_prefetch(Wt + (size_t)(k + 1) * 4096, 0, 1);
    asm volatile("s_wait_asynccnt 0x1" ::: "memory");  // tap k's tile landed
    __syncthreads();                             // all waves' chunks of tap k ready
    compute(k, k & 1);
    __syncthreads();                             // done reading before buf reuse
  }
  asm volatile("s_wait_asynccnt 0x0" ::: "memory");
  __syncthreads();
  compute(TAPS - 1, (TAPS - 1) & 1);

  // C/D 16x16 f32 layout: vgpr v -> row 8*half+v, lane%16 -> col
  const int col = nt * 16 + (lane & 15);
#pragma unroll
  for (int v = 0; v < 8; ++v) {
    int r = row0 + mt * 16 + hf * 8 + v;
    if (r < M) {
      float val = acc[v];
      if (base) val += base[(size_t)r * 64 + col];
      out[(size_t)r * 64 + col] = val;
    }
  }
}

// ---------------------------------------------------------------------------
// Prediction head: BNReLU(x) @ W[64,20] + bias  (tiny; scalar kernel)
// ---------------------------------------------------------------------------
__global__ __launch_bounds__(256) void head_kernel(const float* __restrict__ x, int N,
                                                   const float* __restrict__ ss,
                                                   const float* __restrict__ Wl,
                                                   const float* __restrict__ bias,
                                                   float* __restrict__ out) {
  int i = blockIdx.x * 256 + threadIdx.x;
  if (i >= N * 20) return;
  int s = i / 20, c = i % 20;
  float accv = bias[c];
  const float* xr = x + (size_t)s * 64;
#pragma unroll 8
  for (int k = 0; k < 64; ++k) {
    float v = fmaxf(xr[k] * ss[k] + ss[64 + k], 0.f);
    accv += v * Wl[k * 20 + c];
  }
  out[i] = accv;
}

// ---------------------------------------------------------------------------
// Host orchestration
// ---------------------------------------------------------------------------
extern "C" void kernel_launch(void* const* d_in, const int* in_sizes, int n_in,
                              void* d_out, int out_size, void* d_ws, size_t ws_size,
                              hipStream_t stream) {
  (void)in_sizes; (void)n_in; (void)out_size; (void)ws_size;
  static const int NS[6] = {100000, 50000, 25000, 12500, 6250, 3125};

  int p = 0;
  auto nf = [&]() { return (const float*)d_in[p++]; };
  auto ni = [&]() { return (const int*)d_in[p++]; };

  struct ResP { const float *g1, *b1, *W1, *g2, *b2, *W2; };
  auto rd_res = [&]() {
    ResP r; r.g1 = nf(); r.b1 = nf(); r.W1 = nf();
    r.g2 = nf(); r.b2 = nf(); r.W2 = nf(); return r;
  };

  const float* feats = nf();
  ResP flat[2]; flat[0] = rd_res(); flat[1] = rd_res();
  struct DownP { const float *g, *b, *Wd; ResP res[2]; } down[5];
  for (int i = 0; i < 5; ++i) {
    down[i].g = nf(); down[i].b = nf(); down[i].Wd = nf();
    down[i].res[0] = rd_res(); down[i].res[1] = rd_res();
  }
  struct UpP { const float *g, *b, *Wd; ResP res[2];
               const float *lg, *lb, *lW, *lbias; } up[5];
  for (int i = 0; i < 5; ++i) {
    up[i].g = nf(); up[i].b = nf(); up[i].Wd = nf();
    up[i].res[0] = rd_res(); up[i].res[1] = rd_res();
    up[i].lg = up[i].lb = up[i].lW = up[i].lbias = nullptr;
    if (i >= 3) { up[i].lg = nf(); up[i].lb = nf(); up[i].lW = nf(); up[i].lbias = nf(); }
  }
  const int* nbrs[6]; for (int i = 0; i < 6; ++i) nbrs[i] = ni();
  const int* dnbr[5]; for (int i = 0; i < 5; ++i) dnbr[i] = ni();
  const int* upar[5]; for (int i = 0; i < 5; ++i) upar[i] = ni();
  const int* uoff[5]; for (int i = 0; i < 5; ++i) uoff[i] = ni();

  // ---- workspace carving ----
  char* wsb = (char*)d_ws;
  size_t off = 0;
  auto carve = [&](size_t bytes) -> void* {
    off = (off + 255) & ~(size_t)255;
    void* r = wsb + off; off += bytes; return r;
  };
  float* stats = (float*)carve(128 * sizeof(float));
  float* ssbuf = (float*)carve(128 * sizeof(float));
  unsigned short* wbuf = (unsigned short*)carve(27 * 4096 * sizeof(unsigned short));
  unsigned short* ybuf =
      (unsigned short*)carve((size_t)(NS[0] + 1) * 64 * sizeof(unsigned short));
  float* xs[6];
  for (int l = 0; l < 6; ++l) xs[l] = (float*)carve((size_t)NS[l] * 64 * sizeof(float));
  float* tbuf = (float*)carve((size_t)NS[0] * 64 * sizeof(float));
  int* upidx  = (int*)carve((size_t)NS[0] * 8 * sizeof(int));

  // stats + fused BNReLU->bf16 staging (with zero pad row at index N)
  auto prep = [&](const float* x, int N, const float* g, const float* b) {
    zero_stats<<<1, 128, 0, stream>>>(stats);
    bn_sums<<<256, 256, 0, stream>>>(x, N, stats);
    bn_fin<<<1, 64, 0, stream>>>(stats, g, b, N, ssbuf);
    bnrelu_bf16<<<((N + 1) * 64 + 255) / 256, 256, 0, stream>>>(x, N, ssbuf, ybuf);
  };
  auto conv27 = [&](int Nin, const int* idx, int M, const float* Wf,
                    const float* base, float* out) {
    cvt_w<<<(27 * 4096 + 255) / 256, 256, 0, stream>>>(Wf, 27, wbuf);
    gather_gemm_wmma<27><<<(M + 31) / 32, 256, 0, stream>>>(
        ybuf, Nin, idx, M, wbuf, base, out);
  };
  auto conv8 = [&](int Nin, const int* idx, int M, const float* Wf,
                   const float* base, float* out) {
    cvt_w<<<(8 * 4096 + 255) / 256, 256, 0, stream>>>(Wf, 8, wbuf);
    gather_gemm_wmma<8><<<(M + 31) / 32, 256, 0, stream>>>(
        ybuf, Nin, idx, M, wbuf, base, out);
  };
  auto res_block = [&](float* x, int N, const int* nbr, const ResP& r) {
    prep(x, N, r.g1, r.b1);
    conv27(N, nbr, N, r.W1, nullptr, tbuf);
    prep(tbuf, N, r.g2, r.b2);
    conv27(N, nbr, N, r.W2, x, x);                // x += conv(bnrelu(t)) in place
  };

  // ---- forward ----
  hipMemcpyAsync(xs[0], feats, (size_t)NS[0] * 64 * sizeof(float),
                 hipMemcpyDeviceToDevice, stream);
  res_block(xs[0], NS[0], nbrs[0], flat[0]);
  res_block(xs[0], NS[0], nbrs[0], flat[1]);

  for (int l = 0; l < 5; ++l) {                   // down path: x2..x6
    prep(xs[l], NS[l], down[l].g, down[l].b);
    conv8(NS[l], dnbr[l], NS[l + 1], down[l].Wd, nullptr, xs[l + 1]);
    res_block(xs[l + 1], NS[l + 1], nbrs[l + 1], down[l].res[0]);
    res_block(xs[l + 1], NS[l + 1], nbrs[l + 1], down[l].res[1]);
  }

  float* xup  = xs[5];
  float* outf = (float*)d_out;
  for (int i = 0; i < 5; ++i) {                   // up path: levels 5..1
    int lvl = 4 - i;
    int M = NS[lvl], Ncoarse = NS[lvl + 1];
    prep(xup, Ncoarse, up[i].g, up[i].b);
    build_up_idx<<<(M + 255) / 256, 256, 0, stream>>>(upar[i], uoff[i], M, Ncoarse, upidx);
    conv8(Ncoarse, upidx, M, up[i].Wd, xs[lvl], xs[lvl]);       // skip + deconv
    res_block(xs[lvl], M, nbrs[lvl], up[i].res[0]);
    res_block(xs[lvl], M, nbrs[lvl], up[i].res[1]);
    xup = xs[lvl];
    if (up[i].lW) {                               // pred heads at levels 2 and 1
      zero_stats<<<1, 128, 0, stream>>>(stats);
      bn_sums<<<256, 256, 0, stream>>>(xup, M, stats);
      bn_fin<<<1, 64, 0, stream>>>(stats, up[i].lg, up[i].lb, M, ssbuf);
      float* dst = (i == 4) ? outf : outf + (size_t)NS[0] * 20;
      head_kernel<<<(M * 20 + 255) / 256, 256, 0, stream>>>(
          xup, M, ssbuf, up[i].lW, up[i].lbias, dst);
    }
  }
}
